// RNN_74534862455210
// MI455X (gfx1250) — matrix operand order repaired and checked by reading, not compiled, not run on hardware
//
#include <hip/hip_runtime.h>
#include <hip/hip_bf16.h>

// ---------------------------------------------------------------------------
// 2-layer RNN for MI455X (gfx1250).
//   B=32, T=1024, I=512, H=1024, L=2
// Plan:
//   1. convert x -> [T,B,I] bf16, weights -> bf16
//   2. xi = A @ W^T + bias (big WMMA GEMM, fp32 out)      [per layer]
//   3. persistent scan kernel: h_t = xi_t + h_{t-1} @ Wh^T, Wh slice held
//      in registers per wave, global barrier between steps [per layer]
// ---------------------------------------------------------------------------

typedef __attribute__((ext_vector_type(16))) __bf16 v16bf;
typedef __attribute__((ext_vector_type(8)))  __bf16 v8bf;
typedef __attribute__((ext_vector_type(8)))  float  v8f;

#define RNN_B 32
#define RNN_T 1024
#define RNN_I 512
#define RNN_H 1024
#define RNN_BH (RNN_B * RNN_H)
#define SCAN_NWG 64   // H/16 column slices

__device__ __forceinline__ __bf16 to_bf16(float f) { return (__bf16)f; }

__device__ __forceinline__ v8f bf16_wmma(v16bf a, v16bf b, v8f c) {
  // v_wmma_f32_16x16x32_bf16: D = A(16x32) x B(32x16) + C(16x16 f32)
  return __builtin_amdgcn_wmma_f32_16x16x32_bf16(
      /*neg_a=*/false, a, /*neg_b=*/false, b,
      /*c_mod=*/(short)0, c, /*reuse_a=*/false, /*reuse_b=*/false);
}

// A-matrix 16x32 bf16 tile from row-major [*, lda] memory.
// Contiguous-k layout (not ISA table order; B matches): lanes 0-15 row M=lane hold K 0..15 (v0-7);
//             lanes 16-31 row M=lane-16 hold K 16..31.
__device__ __forceinline__ v16bf load_a_tile(const __bf16* __restrict__ A,
                                             int lda, int m0, int k0, int lane) {
  const int r    = lane & 15;
  const int ksel = (lane < 16) ? 0 : 16;
  const __bf16* p = A + (size_t)(m0 + r) * lda + (k0 + ksel);
  v8bf lo = *(const v8bf*)(p);
  v8bf hi = *(const v8bf*)(p + 8);
  v16bf out;
#pragma unroll
  for (int i = 0; i < 8; ++i) { out[i] = lo[i]; out[8 + i] = hi[i]; }
  return out;
}

// B-matrix 32x16 bf16 tile, where B[k][n] = W[n][k], W row-major [*, ldw].
// Contiguous-k layout (same as A, not ISA table order): lanes 0-15 col N=lane hold K 0..15 packed in v0-7;
//             lanes 16-31 col N=lane-16 hold K 16..31.
__device__ __forceinline__ v16bf load_b_tile(const __bf16* __restrict__ W,
                                             int ldw, int n0, int k0, int lane) {
  const int c    = lane & 15;
  const int koff = (lane < 16) ? 0 : 16;
  return *(const v16bf*)(W + (size_t)(n0 + c) * ldw + (k0 + koff));  // 32B load
}

// ---------------------------------------------------------------------------
// Elementwise converts
// ---------------------------------------------------------------------------
__global__ void f2bf_kernel(const float* __restrict__ in, __bf16* __restrict__ out,
                            size_t n) {
  size_t i = (size_t)blockIdx.x * blockDim.x + threadIdx.x;
  if (i < n) out[i] = to_bf16(in[i]);
}

// x [B,T,I] fp32 -> xbf [T,B,I] bf16 (coalesced along I on both sides)
__global__ void transpose_x_kernel(const float* __restrict__ x,
                                   __bf16* __restrict__ xbf) {
  size_t i = (size_t)blockIdx.x * blockDim.x + threadIdx.x;
  const size_t total = (size_t)RNN_B * RNN_T * RNN_I;
  if (i >= total) return;
  int ii = (int)(i % RNN_I);
  size_t r = i / RNN_I;
  int b = (int)(r % RNN_B);
  int t = (int)(r / RNN_B);
  xbf[i] = to_bf16(x[((size_t)b * RNN_T + t) * RNN_I + ii]);
}

// ---------------------------------------------------------------------------
// C[m,n] = sum_k A[m,k]*W[n,k] + bias[n]      (A,W bf16 row-major; C fp32)
// Block: 256 threads = 8 waves; wave w covers 64 columns; block covers 16 rows
// x 512 columns. Grid: (M/16, N/512).
// ---------------------------------------------------------------------------
__global__ __launch_bounds__(256) void gemm_bias_kernel(
    const __bf16* __restrict__ A, const __bf16* __restrict__ W,
    const float* __restrict__ bias, float* __restrict__ C,
    int M, int N, int K) {
  const int lane = threadIdx.x & 31;
  const int wave = threadIdx.x >> 5;
  const int m0 = blockIdx.x * 16;
  const int n0 = blockIdx.y * 512 + wave * 64;

  v8f acc[4] = {v8f{}, v8f{}, v8f{}, v8f{}};
  for (int k0 = 0; k0 < K; k0 += 32) {
    v16bf a = load_a_tile(A, K, m0, k0, lane);
#pragma unroll
    for (int j = 0; j < 4; ++j) {
      v16bf bt = load_b_tile(W, K, n0 + 16 * j, k0, lane);
      acc[j] = bf16_wmma(a, bt, acc[j]);
    }
  }
  // C/D layout: VGPR r -> M=r (lanes 0-15) or M=r+8 (lanes 16-31); N=lane%16
  const int rowoff = (lane < 16) ? 0 : 8;
  const int c = lane & 15;
#pragma unroll
  for (int j = 0; j < 4; ++j) {
    const int col = n0 + 16 * j + c;
    const float bv = bias[col];
#pragma unroll
    for (int r = 0; r < 8; ++r)
      C[(size_t)(m0 + r + rowoff) * N + col] = acc[j][r] + bv;
  }
}

// ---------------------------------------------------------------------------
// Persistent scan: h_t = xi_t + h_{t-1} @ Wh^T, T steps, global barrier/step.
// Grid: 64 blocks (16 output columns each) x 256 threads (8 waves).
// Wave w owns K-chunk [w*128, w*128+128): its 4 B-tiles of Wh live in
// registers for the whole kernel. Per step: 8 WMMAs/wave, LDS reduction
// across waves, bf16 h_t write, atomic global barrier.
// ring==0: h written to hs + t*BH (layer 0, feeds layer-1 GEMM)
// ring==1: h ping-pongs hs + (t&1)*BH (layer 1); out_final gets fp32 h_{T-1}
// ---------------------------------------------------------------------------
__global__ __launch_bounds__(256) void scan_kernel(
    const float* __restrict__ xi,      // [T,B,H] fp32
    const __bf16* __restrict__ Whl,    // [H,H] bf16, row-major (j,k)
    const __bf16* __restrict__ h_init, // [B,H] bf16
    __bf16* __restrict__ hs,           // ring==0: [T,B,H] ; ring==1: [2,B,H]
    int ring,
    float* __restrict__ out_final,     // [B,H] fp32 or nullptr
    unsigned* __restrict__ bar) {
  const int lane = threadIdx.x & 31;
  const int wave = threadIdx.x >> 5;
  const int tid  = threadIdx.x;
  const int j0   = blockIdx.x * 16;

  __shared__ float red[8 * 512];  // 16 KB partials: [wave][tile][r][lane]

  // Register-resident Wh slice for this wave's K-chunk (32 VGPRs).
  const int kbase = wave * 128;
  v16bf breg[4];
#pragma unroll
  for (int s = 0; s < 4; ++s)
    breg[s] = load_b_tile(Whl, RNN_H, j0, kbase + 32 * s, lane);

  for (int t = 0; t < RNN_T; ++t) {
    const __bf16* hprev =
        (t == 0) ? h_init
                 : hs + (size_t)(ring ? ((t - 1) & 1) : (t - 1)) * RNN_BH;

    v8f acc0 = v8f{}, acc1 = v8f{};
#pragma unroll
    for (int s = 0; s < 4; ++s) {
      const int k0 = kbase + 32 * s;
      v16bf a0 = load_a_tile(hprev, RNN_H, 0, k0, lane);
      acc0 = bf16_wmma(a0, breg[s], acc0);
      v16bf a1 = load_a_tile(hprev, RNN_H, 16, k0, lane);
      acc1 = bf16_wmma(a1, breg[s], acc1);
    }
#pragma unroll
    for (int r = 0; r < 8; ++r) {
      red[wave * 512 + 0 * 256 + r * 32 + lane] = acc0[r];
      red[wave * 512 + 1 * 256 + r * 32 + lane] = acc1[r];
    }
    __syncthreads();

    // 256 threads reduce 512 outputs (2 each), add xi, store bf16 h_t.
    __bf16* hw = hs + (size_t)(ring ? (t & 1) : t) * RNN_BH;
#pragma unroll
    for (int q = 0; q < 2; ++q) {
      const int idx  = tid + q * 256;       // tile*256 + r*32 + l2
      const int tile = idx >> 8;
      const int r    = (idx >> 5) & 7;
      const int l2   = idx & 31;
      float sum = 0.f;
#pragma unroll
      for (int w = 0; w < 8; ++w) sum += red[w * 512 + idx];
      const int bb = tile * 16 + ((l2 < 16) ? 0 : 8) + r;
      const int jj = j0 + (l2 & 15);
      sum += xi[((size_t)t * RNN_B + bb) * RNN_H + jj];
      hw[(size_t)bb * RNN_H + jj] = to_bf16(sum);
      if (out_final != nullptr && t == RNN_T - 1)
        out_final[(size_t)bb * RNN_H + jj] = sum;
    }
    __threadfence();      // make h_t visible agent-wide (release side)
    __syncthreads();      // also protects `red` reuse next iteration

    // device-wide barrier: 64 co-resident blocks, monotonic counter
    if (tid == 0) {
      atomicAdd(bar, 1u);
      const unsigned target = (unsigned)(t + 1) * SCAN_NWG;
      while (__hip_atomic_load(bar, __ATOMIC_RELAXED,
                               __HIP_MEMORY_SCOPE_AGENT) < target) { }
    }
    __syncthreads();
    __builtin_amdgcn_fence(__ATOMIC_ACQUIRE, "agent");  // invalidate for h_t reads
  }
}

// ---------------------------------------------------------------------------
extern "C" void kernel_launch(void* const* d_in, const int* in_sizes, int n_in,
                              void* d_out, int out_size, void* d_ws, size_t ws_size,
                              hipStream_t stream) {
  (void)in_sizes; (void)n_in; (void)out_size; (void)ws_size;
  const float* x    = (const float*)d_in[0];  // [B,T,I]
  const float* Wi0  = (const float*)d_in[1];  // [H,I]
  const float* Wi1  = (const float*)d_in[2];  // [H,H]
  const float* Wh   = (const float*)d_in[3];  // [L,H,H]
  const float* bias = (const float*)d_in[4];  // [L,H]
  float* out = (float*)d_out;                 // [B,H]

  // Workspace carve-up (~242 MB total)
  char* w = (char*)d_ws;
  float*  xi   = (float*)w;  w += (size_t)RNN_T * RNN_B * RNN_H * 4;  // 128 MB
  __bf16* xbf  = (__bf16*)w; w += (size_t)RNN_T * RNN_B * RNN_I * 2;  //  32 MB
  __bf16* hs0  = (__bf16*)w; w += (size_t)RNN_T * RNN_B * RNN_H * 2;  //  64 MB
  __bf16* wi0b = (__bf16*)w; w += (size_t)RNN_H * RNN_I * 2;
  __bf16* wi1b = (__bf16*)w; w += (size_t)RNN_H * RNN_H * 2;
  __bf16* whb  = (__bf16*)w; w += (size_t)2 * RNN_H * RNN_H * 2;
  __bf16* h0   = (__bf16*)w; w += (size_t)RNN_BH * 2;
  __bf16* pp   = (__bf16*)w; w += (size_t)2 * RNN_BH * 2;
  unsigned* bar = (unsigned*)w;

  hipMemsetAsync(h0, 0, (size_t)RNN_BH * 2, stream);  // bf16 zeros == 0x0000

  // Converts
  const size_t nx = (size_t)RNN_B * RNN_T * RNN_I;
  transpose_x_kernel<<<(unsigned)((nx + 255) / 256), 256, 0, stream>>>(x, xbf);
  f2bf_kernel<<<(RNN_H * RNN_I + 255) / 256, 256, 0, stream>>>(Wi0, wi0b,
                                                  (size_t)RNN_H * RNN_I);
  f2bf_kernel<<<(RNN_H * RNN_H + 255) / 256, 256, 0, stream>>>(Wi1, wi1b,
                                                  (size_t)RNN_H * RNN_H);
  f2bf_kernel<<<(2 * RNN_H * RNN_H + 255) / 256, 256, 0, stream>>>(Wh, whb,
                                                  (size_t)2 * RNN_H * RNN_H);

  const int M = RNN_T * RNN_B;  // 32768
  dim3 ggrid(M / 16, RNN_H / 512);

  // Layer 0: projection + scan (h written into hs0[t] for layer-1 GEMM)
  gemm_bias_kernel<<<ggrid, 256, 0, stream>>>(xbf, wi0b, bias, xi,
                                              M, RNN_H, RNN_I);
  hipMemsetAsync(bar, 0, sizeof(unsigned), stream);
  scan_kernel<<<SCAN_NWG, 256, 0, stream>>>(xi, whb, h0, hs0, /*ring=*/0,
                                            nullptr, bar);

  // Layer 1: projection over hs0, scan seeded with h_{T-1} of layer 0
  gemm_bias_kernel<<<ggrid, 256, 0, stream>>>(hs0, wi1b, bias + RNN_H, xi,
                                              M, RNN_H, RNN_H);
  hipMemsetAsync(bar, 0, sizeof(unsigned), stream);
  scan_kernel<<<SCAN_NWG, 256, 0, stream>>>(
      xi, whb + (size_t)RNN_H * RNN_H,
      hs0 + (size_t)(RNN_T - 1) * RNN_BH, pp, /*ring=*/1, out, bar);
}
